// Model_83734682403185
// MI455X (gfx1250) — matrix operand order repaired and checked by reading, not compiled, not run on hardware
//
#include <hip/hip_runtime.h>

typedef __attribute__((ext_vector_type(16))) __bf16        v16bf;
typedef __attribute__((ext_vector_type(8)))  float         v8f;
typedef __attribute__((ext_vector_type(8)))  unsigned int  v8u;

#define BB 64
#define SS 512
#define HH 768
#define MM 512
#define TT 26
#define NROWS (BB*SS)     // 32768
#define TPAD 32
#define KCH 32
#define LSTR 40           // LDS row stride (ushorts): 80B rows -> 16B aligned, conflict-free 16B lane loads

// ---- fp32 -> bf16 (round to nearest even) ----
__device__ __forceinline__ unsigned short f2bf(float f) {
  unsigned int u = __float_as_uint(f);
  u += 0x7FFFu + ((u >> 16) & 1u);
  return (unsigned short)(u >> 16);
}
__device__ __forceinline__ unsigned int pk2(float a, float b) {
  return (unsigned int)f2bf(a) | ((unsigned int)f2bf(b) << 16);
}
__device__ __forceinline__ uint4 pk8(float4 a, float4 b) {
  uint4 v;
  v.x = pk2(a.x, a.y); v.y = pk2(a.z, a.w);
  v.z = pk2(b.x, b.y); v.w = pk2(b.z, b.w);
  return v;
}

// ---- WMMA fragment builders (CDNA5 ISA 7.12.2), two ds_load_b128 each ----
// A 16x32 bf16: lanes<16 hold K0..7 (V0..3) + K16..23 (V4..7); lanes>=16: +8.
__device__ __forceinline__ v16bf frag_a(const unsigned short* ls, int row0,
                                        int lanelo, int lanehi) {
  const unsigned short* base = ls + (row0 + lanelo) * LSTR + (lanehi ? 8 : 0);
  uint4 lo = *(const uint4*)(base);        // 16B: K 0..7   (or 8..15)
  uint4 hi = *(const uint4*)(base + 16);   // 16B: K 16..23 (or 24..31)
  v8u u;
  u[0] = lo.x; u[1] = lo.y; u[2] = lo.z; u[3] = lo.w;
  u[4] = hi.x; u[5] = hi.y; u[6] = hi.z; u[7] = hi.w;
  return __builtin_bit_cast(v16bf, u);
}
// B 32x16 bf16 stored [n][k]: lanes<16 hold K0..7 + K16..23; lanes>=16: +8.
__device__ __forceinline__ v16bf frag_b(const unsigned short* ls, int col0,
                                        int lanelo, int lanehi) {
  const unsigned short* base = ls + (col0 + lanelo) * LSTR + (lanehi ? 8 : 0);
  uint4 lo = *(const uint4*)(base);
  uint4 hi = *(const uint4*)(base + 16);
  v8u u;
  u[0] = lo.x; u[1] = lo.y; u[2] = lo.z; u[3] = lo.w;
  u[4] = hi.x; u[5] = hi.y; u[6] = hi.z; u[7] = hi.w;
  return __builtin_bit_cast(v16bf, u);
}

// ========== GEMM1: h = hidden(32768x768) @ W1^T + b1 -> bf16 ==========
// Block tile 128x128, 8 waves of 64x32 (8 WMMAs / wave / k-step), K chunk 32.
__global__ __launch_bounds__(256) void gemm1_kernel(const float* __restrict__ hidden,
                                                    const float* __restrict__ W1,
                                                    const float* __restrict__ b1,
                                                    unsigned short* __restrict__ hout) {
  __shared__ unsigned short lsA[128 * LSTR];  // 10240 B
  __shared__ unsigned short lsB[128 * LSTR];  // 10240 B
  const int tid = threadIdx.x;
  const int lane = tid & 31, wv = tid >> 5;
  const int lanelo = lane & 15, lanehi = lane >> 4;
  const int row0 = (int)blockIdx.x * 128;
  const int col0 = (int)blockIdx.y * 128;
  const int wr = (wv & 1) * 64;
  const int wc = (wv >> 1) * 32;

  v8f acc[4][2] = {};

  // both A and B staging: thread -> row/col = tid>>1, 16 k's at (tid&1)*16
  const int sr = tid >> 1, sk = (tid & 1) * 16;

  for (int kb = 0; kb < HH; kb += KCH) {
    {
      const float4* gp = (const float4*)(hidden + (size_t)(row0 + sr) * HH + kb + sk);
      if (kb + KCH < HH) __builtin_prefetch(gp + 8, 0, 3);
      float4 x0 = gp[0], x1 = gp[1], x2 = gp[2], x3 = gp[3];
      uint4* lp = (uint4*)&lsA[sr * LSTR + sk];
      lp[0] = pk8(x0, x1);
      lp[1] = pk8(x2, x3);
    }
    {
      const float4* gp = (const float4*)(W1 + (size_t)(col0 + sr) * HH + kb + sk);
      float4 x0 = gp[0], x1 = gp[1], x2 = gp[2], x3 = gp[3];
      uint4* lp = (uint4*)&lsB[sr * LSTR + sk];
      lp[0] = pk8(x0, x1);
      lp[1] = pk8(x2, x3);
    }
    __syncthreads();
    v16bf bf0 = frag_b(lsB, wc, lanelo, lanehi);
    v16bf bf1 = frag_b(lsB, wc + 16, lanelo, lanehi);
#pragma unroll
    for (int mi = 0; mi < 4; ++mi) {
      v16bf af = frag_a(lsA, wr + mi * 16, lanelo, lanehi);
      acc[mi][0] = __builtin_amdgcn_wmma_f32_16x16x32_bf16(false, af, false, bf0, (short)0, acc[mi][0], false, false);
      acc[mi][1] = __builtin_amdgcn_wmma_f32_16x16x32_bf16(false, af, false, bf1, (short)0, acc[mi][1], false, false);
    }
    __syncthreads();
  }

  // C element e -> row = e + 8*lanehi, col = lanelo (per 16x16 tile)
#pragma unroll
  for (int ni = 0; ni < 2; ++ni) {
    const int col = col0 + wc + ni * 16 + lanelo;
    const float bias = b1[col];
#pragma unroll
    for (int mi = 0; mi < 4; ++mi) {
#pragma unroll
      for (int e = 0; e < 8; ++e) {
        const int row = row0 + wr + mi * 16 + e + 8 * lanehi;
        hout[(size_t)row * MM + col] = f2bf(acc[mi][ni][e] + bias);
      }
    }
  }
}

// ========== GEMM2: emis = h(32768x512,bf16) @ W2^T + b2 -> f32 (T padded to 32) ==========
__global__ __launch_bounds__(256) void gemm2_kernel(const unsigned short* __restrict__ h,
                                                    const float* __restrict__ W2,
                                                    const float* __restrict__ b2,
                                                    float* __restrict__ emis) {
  __shared__ unsigned short lsA[128 * LSTR];
  __shared__ unsigned short lsB[32 * LSTR];
  const int tid = threadIdx.x;
  const int lane = tid & 31, wv = tid >> 5;
  const int lanelo = lane & 15, lanehi = lane >> 4;
  const int row0 = (int)blockIdx.x * 128;
  const int wr = wv * 16;

  v8f acc[2] = {};

  const int ar = tid >> 1, ak = (tid & 1) * 16;  // A: 32B per thread (16 bf16)
  const int bn = tid >> 3, bk = (tid & 7) * 4;   // B: 4 f32 per thread

  for (int kb = 0; kb < MM; kb += KCH) {
    {
      // pure byte copy of bf16 tile: use CDNA5 async global->LDS DMA (ASYNCcnt)
      const unsigned short* gsrc = h + (size_t)(row0 + ar) * MM + kb + ak;  // 32B aligned
      unsigned lds_off = (unsigned)(size_t)&lsA[ar * LSTR + ak];            // 16B aligned
      unsigned long long ga = (unsigned long long)(size_t)gsrc;
      asm volatile("global_load_async_to_lds_b128 %0, %1, off\n\t"
                   "global_load_async_to_lds_b128 %0, %1, off offset:16"
                   :: "v"(lds_off), "v"(ga) : "memory");
    }
    {
      float4 v = make_float4(0.f, 0.f, 0.f, 0.f);
      if (bn < TT) v = *(const float4*)(W2 + (size_t)bn * MM + kb + bk);
      unsigned int* lp = (unsigned int*)&lsB[bn * LSTR + bk];
      lp[0] = pk2(v.x, v.y);
      lp[1] = pk2(v.z, v.w);
    }
    asm volatile("s_wait_asynccnt 0x0" ::: "memory");
    __syncthreads();
    v16bf af  = frag_a(lsA, wr, lanelo, lanehi);
    v16bf bf0 = frag_b(lsB, 0, lanelo, lanehi);
    v16bf bf1 = frag_b(lsB, 16, lanelo, lanehi);
    acc[0] = __builtin_amdgcn_wmma_f32_16x16x32_bf16(false, af, false, bf0, (short)0, acc[0], false, false);
    acc[1] = __builtin_amdgcn_wmma_f32_16x16x32_bf16(false, af, false, bf1, (short)0, acc[1], false, false);
    __syncthreads();
  }

#pragma unroll
  for (int ni = 0; ni < 2; ++ni) {
    const int col = ni * 16 + lanelo;
    const float bias = (col < TT) ? b2[col] : 0.0f;
#pragma unroll
    for (int e = 0; e < 8; ++e) {
      const int row = row0 + wr + e + 8 * lanehi;
      emis[(size_t)row * TPAD + col] = acc[ni][e] + bias;
    }
  }
}

// ========== CRF: one wave32 per batch; exp(trans) cached in registers ==========
__global__ __launch_bounds__(32) void crf_kernel(const float* __restrict__ emis,
                                                 const int* __restrict__ labels,
                                                 const int* __restrict__ lengths,
                                                 const float* __restrict__ start_trans,
                                                 const float* __restrict__ end_trans,
                                                 const float* __restrict__ trans,
                                                 float* __restrict__ llh) {
  __shared__ float tS[TT * 32];  // trans padded to stride 32 (bank-conflict free)
  const int b = blockIdx.x;
  const int j = threadIdx.x;
  for (int i = 0; i < TT; ++i)
    tS[i * 32 + j] = (j < TT) ? trans[i * TT + j] : 0.0f;
  __syncthreads();

  // lane j caches column j of exp(trans): E[i] = exp(trans[i][j])
  float E[TT];
#pragma unroll
  for (int i = 0; i < TT; ++i) E[i] = __expf(tS[i * 32 + j]);

  const int L = lengths[b];
  const int* lab = labels + b * SS;
  const float* em = emis + (size_t)b * SS * TPAD;
  const float NEG = -1.0e30f;

  // ---- numerator (lane-parallel over s) ----
  float np = 0.0f;
  for (int s = 1 + j; s < L; s += 32) {
    int l0 = lab[s - 1], l1 = lab[s];
    np += tS[l0 * 32 + l1] + em[(size_t)s * TPAD + l1];
  }
#pragma unroll
  for (int o = 16; o > 0; o >>= 1) np += __shfl_xor(np, o, 32);
  const int lab0 = lab[0];
  const int labE = lab[L - 1];
  float numer = np + start_trans[lab0] + em[lab0] + end_trans[labE];

  // ---- forward recursion: alpha band is narrow (|trans|~0.4, |emit|~1), so
  //      lane-0 alpha is a safe normalizer; 1 exp + 1 log + 26 shfl-FMA per step.
  float alpha = (j < TT) ? (start_trans[j] + em[j]) : NEG;
  for (int s = 1; s < L; ++s) {
    float m = __shfl(alpha, 0, 32);
    float ea = __expf(alpha - m);          // lanes >= TT: exp(NEG-m) == 0
    float ssum = 0.0f;
#pragma unroll
    for (int i = 0; i < TT; ++i)
      ssum = fmaf(__shfl(ea, i, 32), E[i], ssum);
    float nxt = m + __logf(ssum) + em[(size_t)s * TPAD + j];
    alpha = (j < TT) ? nxt : NEG;
  }

  // ---- denominator = logsumexp(alpha + end_trans) (exact two-pass, runs once) ----
  float v = (j < TT) ? (alpha + end_trans[j]) : NEG;
  float m = v;
#pragma unroll
  for (int o = 16; o > 0; o >>= 1) m = fmaxf(m, __shfl_xor(m, o, 32));
  float se = __expf(v - m);
#pragma unroll
  for (int o = 16; o > 0; o >>= 1) se += __shfl_xor(se, o, 32);
  if (j == 0) llh[b] = numer - (m + __logf(se));
}

// ========== final: out = -mean(llh) ==========
__global__ void final_kernel(const float* __restrict__ llh, float* __restrict__ out) {
  if (threadIdx.x == 0) {
    float s = 0.0f;
    for (int b = 0; b < BB; ++b) s += llh[b];
    out[0] = -s / (float)BB;
  }
}

extern "C" void kernel_launch(void* const* d_in, const int* in_sizes, int n_in,
                              void* d_out, int out_size, void* d_ws, size_t ws_size,
                              hipStream_t stream) {
  const float* hidden      = (const float*)d_in[0];
  const float* W1          = (const float*)d_in[1];
  const float* b1          = (const float*)d_in[2];
  const float* W2          = (const float*)d_in[3];
  const float* b2          = (const float*)d_in[4];
  const float* start_trans = (const float*)d_in[5];
  const float* end_trans   = (const float*)d_in[6];
  const float* trans       = (const float*)d_in[7];
  const int*   labels      = (const int*)d_in[8];
  const int*   lengths     = (const int*)d_in[9];

  char* ws = (char*)d_ws;
  unsigned short* h_bf = (unsigned short*)ws;                         // 32 MB
  float* emis = (float*)(ws + (size_t)NROWS * MM * 2);                //  4 MB
  float* llh  = (float*)(ws + (size_t)NROWS * MM * 2 + (size_t)NROWS * TPAD * 4);

  hipLaunchKernelGGL(gemm1_kernel, dim3(NROWS / 128, MM / 128), dim3(256), 0, stream,
                     hidden, W1, b1, h_bf);
  hipLaunchKernelGGL(gemm2_kernel, dim3(NROWS / 128), dim3(256), 0, stream,
                     h_bf, W2, b2, emis);
  hipLaunchKernelGGL(crf_kernel, dim3(BB), dim3(32), 0, stream,
                     emis, labels, lengths, start_trans, end_trans, trans, llh);
  hipLaunchKernelGGL(final_kernel, dim3(1), dim3(32), 0, stream, llh, (float*)d_out);
}
